// EGNNLayer_64768106824107
// MI455X (gfx1250) — matrix operand order repaired and checked by reading, not compiled, not run on hardware
//
#include <hip/hip_runtime.h>
#include <math.h>

// ---------------------------------------------------------------------------
// EGNN layer for MI455X (gfx1250). Strategy:
//   * msg / dx edge MLPs via v_wmma_f32_16x16x32_f16 (f16 in, f32 accum)
//   * per-node base = h_i @ msg_w1[0:64] + b1 folded into WMMA C-init
//   * ALL WMMA operands (A tiles and packed weights) fetched from LDS:
//     weights (34.8 KB f16) staged once per workgroup
//   * per-wave 16-edge tiles staged through LDS, msum/gdx via ds_add_f32
//   * tiny per-node head MLPs in a scalar VALU kernel
// ---------------------------------------------------------------------------

typedef __attribute__((ext_vector_type(16))) _Float16 v16h;
typedef __attribute__((ext_vector_type(8)))  float    v8f;

union Frag16 { v16h v; uint4 q[2]; };
union Acc8   { v8f  v; float f[8]; };

#define WMMA_F16(A, B, C) \
  __builtin_amdgcn_wmma_f32_16x16x32_f16(false, (A), false, (B), (short)0, (C), false, false)

#define NB 384
#define PB 128

// packed-weight region offsets (in halves) inside d_ws / LDS stage
#define W1P_OFF 0       // 64 x 128
#define W2P_OFF 8192    // 64 x 64
#define T1P_OFF 12288   // 64 x 64
#define T2P_OFF 16384   // 16 x 64
#define WPK_TOT 17408   // total halves (34816 bytes)

static __device__ __forceinline__ void qmul(const float a[4], const float b[4], float o[4]) {
  o[0] = a[0]*b[0] - a[1]*b[1] - a[2]*b[2] - a[3]*b[3];
  o[1] = a[0]*b[1] + a[1]*b[0] + a[2]*b[3] - a[3]*b[2];
  o[2] = a[0]*b[2] - a[1]*b[3] + a[2]*b[0] + a[3]*b[1];
  o[3] = a[0]*b[3] + a[1]*b[2] - a[2]*b[1] + a[3]*b[0];
}

static __device__ __forceinline__ void qrot(const float q[4], const float v[3], float o[3]) {
  const float w = q[0], ux = q[1], uy = q[2], uz = q[3];
  const float tx = 2.f*(uy*v[2] - uz*v[1]);
  const float ty = 2.f*(uz*v[0] - ux*v[2]);
  const float tz = 2.f*(ux*v[1] - uy*v[0]);
  o[0] = v[0] + w*tx + (uy*tz - uz*ty);
  o[1] = v[1] + w*ty + (uz*tx - ux*tz);
  o[2] = v[2] + w*tz + (ux*ty - uy*tx);
}

// ---------------------------------------------------------------------------
// Prepack weights into f16, transposed to [n][k] so B-fragments are two
// contiguous uint4 loads per lane.
//   w1p : 64 x 128  (= msg_w1 rows 64..168 transposed, K-padded with zeros)
//   w2p : 64 x 64   (msg_w2 transposed)
//   t1p : 64 x 64   (tr_w1 transposed)
//   t2p : 16 x 64   (tr_w2 transposed, N-padded 3 -> 16 with zeros)
// ---------------------------------------------------------------------------
__global__ __launch_bounds__(256) void prepack_kernel(
    const float* __restrict__ w1, const float* __restrict__ w2,
    const float* __restrict__ t1, const float* __restrict__ t2,
    _Float16* __restrict__ ws)
{
  const int idx = blockIdx.x * 256 + threadIdx.x;
  if (idx < 8192) {
    const int n = idx >> 7, k = idx & 127;
    ws[W1P_OFF + idx] = (k < 105) ? (_Float16)w1[(64 + k) * 64 + n] : (_Float16)0.f;
  } else if (idx < 12288) {
    const int r = idx - 8192, n = r >> 6, k = r & 63;
    ws[W2P_OFF + r] = (_Float16)w2[k * 64 + n];
  } else if (idx < 16384) {
    const int r = idx - 12288, n = r >> 6, k = r & 63;
    ws[T1P_OFF + r] = (_Float16)t1[k * 64 + n];
  } else if (idx < 17408) {
    const int r = idx - 16384, n = r >> 6, k = r & 63;
    ws[T2P_OFF + r] = (n < 3) ? (_Float16)t2[k * 3 + n] : (_Float16)0.f;
  }
}

// ---------------------------------------------------------------------------
// Edge kernel: one workgroup (128 thr = 4 waves) per node i. Each wave owns
// 8 tiles of 16 neighbors; per tile: feat build -> msg MLP (WMMA) -> mask ->
// msum accumulate -> dx MLP (WMMA) -> rotate -> gdx accumulate.
// ---------------------------------------------------------------------------
__global__ __launch_bounds__(128) void egnn_edge_kernel(
    const float* __restrict__ fq, const float* __restrict__ fx,
    const float* __restrict__ h,  const float* __restrict__ eN,
    const int*   __restrict__ nm, const float* __restrict__ ph,
    const float* __restrict__ eP, const float* __restrict__ pq,
    const float* __restrict__ px, const int*   __restrict__ pm,
    const float* __restrict__ msg_w1, const float* __restrict__ msg_b1,
    const float* __restrict__ msg_b2, const float* __restrict__ tr_b1,
    const float* __restrict__ tr_b2,
    const _Float16* __restrict__ wpack,
    float* __restrict__ msum_g, float* __restrict__ gdx_g)
{
  const int bi   = blockIdx.x;
  const int b    = bi / NB;
  const int i    = bi - b * NB;
  const int tid  = threadIdx.x;
  const int wave = tid >> 5;
  const int lane = tid & 31;
  const int m16  = lane & 15;
  const int hi   = lane >> 4;

  __shared__ float s_base[64], s_b2[64], s_tb1[64], s_tb2[4];
  __shared__ float s_msum[64], s_gdx[4];
  __shared__ float s_qi[4], s_xi[4], s_maski;
  __shared__ alignas(16) _Float16 s_w[WPK_TOT];           // 34.8 KB packed weights
  __shared__ alignas(16) _Float16 s_bufA[4][16 * 128];    // feat / m stage
  __shared__ alignas(16) _Float16 s_bufB[4][16 * 64];     // hidden stages
  __shared__ float s_mask[4][16];
  __shared__ float s_dx[4][16][4];

  // stage packed weights into LDS (coalesced b128 copies)
  {
    uint4*       dst4 = (uint4*)s_w;
    const uint4* src4 = (const uint4*)wpack;
    for (int idx = tid; idx < WPK_TOT / 8; idx += 128) dst4[idx] = src4[idx];
  }

  if (tid < 64) {
    // base[n] = msg_b1[n] + h_i @ msg_w1[0:64, n]  (folds h_i block + bias)
    float acc = msg_b1[tid];
    const float* hrow = h + (size_t)bi * 64;
    for (int k = 0; k < 64; ++k) acc += hrow[k] * msg_w1[k * 64 + tid];
    s_base[tid] = acc;
    s_b2[tid]   = msg_b2[tid];
    s_tb1[tid]  = tr_b1[tid];
    s_msum[tid] = 0.f;
  } else if (tid < 68) s_qi[tid - 64] = fq[(size_t)bi * 4 + (tid - 64)];
  else if (tid < 71)   s_xi[tid - 68] = fx[(size_t)bi * 3 + (tid - 68)];
  else if (tid < 74)   s_tb2[tid - 71] = tr_b2[tid - 71];
  else if (tid == 74)  { s_tb2[3] = 0.f; s_maski = (nm[bi] != 0) ? 1.f : 0.f; }
  else if (tid < 79)   s_gdx[tid - 75] = 0.f;
  __syncthreads();

  _Float16* bufA = s_bufA[wave];
  _Float16* bufB = s_bufB[wave];
  const _Float16* sw1p = s_w + W1P_OFF;
  const _Float16* sw2p = s_w + W2P_OFF;
  const _Float16* st1p = s_w + T1P_OFF;
  const _Float16* st2p = s_w + T2P_OFF;
  const int ako = hi * 8;    // A-fragment K offset within 32-chunk
  const int bko = hi * 8;    // B-fragment K offset within 32-chunk

  for (int tt = 0; tt < 8; ++tt) {
    const int tile = tt * 4 + wave;

    // ---- build 16 x 128 f16 feature tile (K: h_j 0..63, e 64..95, geom 96..104, pad) ----
    for (int e16 = 0; e16 < 16; ++e16) {
      const int j = tile * 16 + e16;
      const float* hsrc;
      const float* esrc;
      if (j < NB) {
        hsrc = h  + (size_t)(b * NB + j) * 64;
        esrc = eN + ((size_t)(b * NB + i) * NB + j) * 32;
      } else {
        const int p = j - NB;
        hsrc = ph + (size_t)(b * PB + p) * 64;
        esrc = eP + ((size_t)(b * NB + i) * PB + p) * 32;
      }
      const float2 hv = *(const float2*)(hsrc + 2 * lane);
      bufA[e16 * 128 + 2 * lane    ] = (_Float16)hv.x;
      bufA[e16 * 128 + 2 * lane + 1] = (_Float16)hv.y;
      bufA[e16 * 128 + 64 + lane   ] = (_Float16)esrc[lane];
    }
    if (hi == 0) {
      const int e16 = m16;
      const int j = tile * 16 + e16;
      float qj[4], xj[3], mk;
      if (j < NB) {
        const float* qs = fq + (size_t)(b * NB + j) * 4;
        const float* xs = fx + (size_t)(b * NB + j) * 3;
        qj[0]=qs[0]; qj[1]=qs[1]; qj[2]=qs[2]; qj[3]=qs[3];
        xj[0]=xs[0]; xj[1]=xs[1]; xj[2]=xs[2];
        mk = ((nm[b * NB + j] != 0) && (j != i)) ? s_maski : 0.f;
      } else {
        const int p = j - NB;
        const float* qs = pq + (size_t)(b * PB + p) * 4;
        const float* xs = px + (size_t)(b * PB + p) * 3;
        qj[0]=qs[0]; qj[1]=qs[1]; qj[2]=qs[2]; qj[3]=qs[3];
        xj[0]=xs[0]; xj[1]=xs[1]; xj[2]=xs[2];
        mk = (pm[b * PB + p] != 0) ? s_maski : 0.f;
      }
      const float vec[3] = { s_xi[0]-xj[0], s_xi[1]-xj[1], s_xi[2]-xj[2] };
      const float d2 = vec[0]*vec[0] + vec[1]*vec[1] + vec[2]*vec[2];
      const float qiv[4] = { s_qi[0], s_qi[1], s_qi[2], s_qi[3] };
      const float qdot = fabsf(qiv[0]*qj[0] + qiv[1]*qj[1] + qiv[2]*qj[2] + qiv[3]*qj[3]);
      const float qc[4] = { qj[0], -qj[1], -qj[2], -qj[3] };
      float lx[3]; qrot(qc, vec, lx);
      float lq[4]; qmul(qc, qiv, lq);
      _Float16* fr = bufA + e16 * 128;
      fr[96]=(_Float16)lx[0]; fr[97]=(_Float16)lx[1]; fr[98]=(_Float16)lx[2];
      fr[99]=(_Float16)lq[0]; fr[100]=(_Float16)lq[1];
      fr[101]=(_Float16)lq[2]; fr[102]=(_Float16)lq[3];
      fr[103]=(_Float16)d2;   fr[104]=(_Float16)qdot;
      s_mask[wave][e16] = mk;
    } else {
      _Float16* fr = bufA + m16 * 128;
      for (int k = 105; k < 128; ++k) fr[k] = (_Float16)0.f;
    }
    __syncthreads();

    // ---- msg layer 1: feat[16x128] @ w1p -> hidden, C init = base ----
    Acc8 acc[4];
    for (int nt = 0; nt < 4; ++nt) {
      const float bv = s_base[nt * 16 + m16];
      for (int r = 0; r < 8; ++r) acc[nt].f[r] = bv;
    }
    for (int c = 0; c < 4; ++c) {
      Frag16 a;
      const _Float16* ap = bufA + m16 * 128 + c * 32 + ako;
      a.q[0] = *(const uint4*)(ap);
      a.q[1] = *(const uint4*)(ap + 16);
      for (int nt = 0; nt < 4; ++nt) {
        Frag16 bf;
        const _Float16* bp = sw1p + (nt * 16 + m16) * 128 + c * 32 + bko;
        bf.q[0] = *(const uint4*)(bp);
        bf.q[1] = *(const uint4*)(bp + 16);
        acc[nt].v = WMMA_F16(a.v, bf.v, acc[nt].v);
      }
    }
    for (int nt = 0; nt < 4; ++nt)
      for (int r = 0; r < 8; ++r) {
        const float v = acc[nt].f[r];
        bufB[(r + 8 * hi) * 64 + nt * 16 + m16] = (_Float16)(v > 0.f ? v : 0.f);
      }
    __syncthreads();

    // ---- msg layer 2: hidden[16x64] @ w2p -> m, mask, msum, stage m ----
    Acc8 acc2[4];
    for (int nt = 0; nt < 4; ++nt) {
      const float bv = s_b2[nt * 16 + m16];
      for (int r = 0; r < 8; ++r) acc2[nt].f[r] = bv;
    }
    for (int c = 0; c < 2; ++c) {
      Frag16 a;
      const _Float16* ap = bufB + m16 * 64 + c * 32 + ako;
      a.q[0] = *(const uint4*)(ap);
      a.q[1] = *(const uint4*)(ap + 16);
      for (int nt = 0; nt < 4; ++nt) {
        Frag16 bf;
        const _Float16* bp = sw2p + (nt * 16 + m16) * 64 + c * 32 + bko;
        bf.q[0] = *(const uint4*)(bp);
        bf.q[1] = *(const uint4*)(bp + 16);
        acc2[nt].v = WMMA_F16(a.v, bf.v, acc2[nt].v);
      }
    }
    float maskv[8];
    for (int r = 0; r < 8; ++r) maskv[r] = s_mask[wave][r + 8 * hi];
    for (int nt = 0; nt < 4; ++nt) {
      float s = 0.f;
      for (int r = 0; r < 8; ++r) {
        const float v = acc2[nt].f[r] * maskv[r];
        bufA[(r + 8 * hi) * 64 + nt * 16 + m16] = (_Float16)v;  // masked m
        s += v;
      }
      atomicAdd(&s_msum[nt * 16 + m16], s);
    }
    __syncthreads();

    // ---- dx layer 1: m[16x64] @ t1p -> relu -> hidden2 ----
    Acc8 acc3[4];
    for (int nt = 0; nt < 4; ++nt) {
      const float bv = s_tb1[nt * 16 + m16];
      for (int r = 0; r < 8; ++r) acc3[nt].f[r] = bv;
    }
    for (int c = 0; c < 2; ++c) {
      Frag16 a;
      const _Float16* ap = bufA + m16 * 64 + c * 32 + ako;
      a.q[0] = *(const uint4*)(ap);
      a.q[1] = *(const uint4*)(ap + 16);
      for (int nt = 0; nt < 4; ++nt) {
        Frag16 bf;
        const _Float16* bp = st1p + (nt * 16 + m16) * 64 + c * 32 + bko;
        bf.q[0] = *(const uint4*)(bp);
        bf.q[1] = *(const uint4*)(bp + 16);
        acc3[nt].v = WMMA_F16(a.v, bf.v, acc3[nt].v);
      }
    }
    for (int nt = 0; nt < 4; ++nt)
      for (int r = 0; r < 8; ++r) {
        const float v = acc3[nt].f[r];
        bufB[(r + 8 * hi) * 64 + nt * 16 + m16] = (_Float16)(v > 0.f ? v : 0.f);
      }
    __syncthreads();

    // ---- dx layer 2: hidden2[16x64] @ t2p (64x16, cols 0..2 valid) ----
    Acc8 acc4;
    {
      const float bv = (m16 < 3) ? s_tb2[m16] : 0.f;
      for (int r = 0; r < 8; ++r) acc4.f[r] = bv;
    }
    for (int c = 0; c < 2; ++c) {
      Frag16 a;
      const _Float16* ap = bufB + m16 * 64 + c * 32 + ako;
      a.q[0] = *(const uint4*)(ap);
      a.q[1] = *(const uint4*)(ap + 16);
      Frag16 bf;
      const _Float16* bp = st2p + m16 * 64 + c * 32 + bko;
      bf.q[0] = *(const uint4*)(bp);
      bf.q[1] = *(const uint4*)(bp + 16);
      acc4.v = WMMA_F16(a.v, bf.v, acc4.v);
    }
    if (m16 < 3)
      for (int r = 0; r < 8; ++r)
        s_dx[wave][r + 8 * hi][m16] = acc4.f[r] * maskv[r];
    __syncthreads();

    // ---- rotate dx by normalize(q_j) and accumulate gdx ----
    if (hi == 0) {
      const int e16 = m16;
      const int j = tile * 16 + e16;
      float qj[4];
      if (j < NB) {
        const float* qs = fq + (size_t)(b * NB + j) * 4;
        qj[0]=qs[0]; qj[1]=qs[1]; qj[2]=qs[2]; qj[3]=qs[3];
      } else {
        const float* qs = pq + (size_t)(b * PB + (j - NB)) * 4;
        qj[0]=qs[0]; qj[1]=qs[1]; qj[2]=qs[2]; qj[3]=qs[3];
      }
      float nn = sqrtf(qj[0]*qj[0] + qj[1]*qj[1] + qj[2]*qj[2] + qj[3]*qj[3]);
      nn = fmaxf(nn, 1e-12f);
      const float qn[4] = { qj[0]/nn, qj[1]/nn, qj[2]/nn, qj[3]/nn };
      const float dxv[3] = { s_dx[wave][e16][0], s_dx[wave][e16][1], s_dx[wave][e16][2] };
      float g[3]; qrot(qn, dxv, g);
      atomicAdd(&s_gdx[0], g[0]);
      atomicAdd(&s_gdx[1], g[1]);
      atomicAdd(&s_gdx[2], g[2]);
    }
    __syncthreads();
  }

  if (tid < 64)       msum_g[(size_t)bi * 64 + tid] = s_msum[tid];
  else if (tid < 68)  gdx_g[(size_t)bi * 4 + (tid - 64)] = s_gdx[tid - 64];
}

// ---------------------------------------------------------------------------
// Node kernel: per-node head MLPs + quaternion/torsion/position updates.
// ---------------------------------------------------------------------------
__global__ __launch_bounds__(64) void egnn_node_kernel(
    const float* __restrict__ fq, const float* __restrict__ fx,
    const float* __restrict__ h,
    const int* __restrict__ nm, const int* __restrict__ pm,
    const float* __restrict__ fw1, const float* __restrict__ fb1,
    const float* __restrict__ fw2, const float* __restrict__ fb2,
    const float* __restrict__ qw1, const float* __restrict__ qb1,
    const float* __restrict__ qw2, const float* __restrict__ qb2,
    const float* __restrict__ tw1, const float* __restrict__ tb1,
    const float* __restrict__ tw2, const float* __restrict__ tb2,
    const float* __restrict__ msum_g, const float* __restrict__ gdx_g,
    float* __restrict__ out)
{
  const int bi = blockIdx.x;
  const int b  = bi / NB;
  const int i  = bi - b * NB;
  const int t  = threadIdx.x;

  __shared__ float sx[128];
  __shared__ float hid[64];
  __shared__ float sdq[4];
  __shared__ float stor[14];
  __shared__ int scnt;

  if (t == 0) scnt = 0;
  sx[t]      = h[(size_t)bi * 64 + t];
  sx[64 + t] = msum_g[(size_t)bi * 64 + t];
  int cn = 0;
  for (int k = t; k < NB; k += 64) cn += (nm[b * NB + k] != 0);
  for (int k = t; k < PB; k += 64) cn += (pm[b * PB + k] != 0);
  __syncthreads();
  atomicAdd(&scnt, cn);

  // o = MLP([h, msum])
  float a = fb1[t];
  for (int k = 0; k < 128; ++k) a += sx[k] * fw1[k * 64 + t];
  hid[t] = fmaxf(a, 0.f);
  __syncthreads();
  float o = fb2[t];
  for (int k = 0; k < 64; ++k) o += hid[k] * fw2[k * 64 + t];
  out[16128 + (size_t)bi * 64 + t] = o;
  __syncthreads();

  // dq = MLP(msum)
  a = qb1[t];
  for (int k = 0; k < 64; ++k) a += sx[64 + k] * qw1[k * 64 + t];
  hid[t] = fmaxf(a, 0.f);
  __syncthreads();
  if (t < 4) {
    float d = qb2[t];
    for (int k = 0; k < 64; ++k) d += hid[k] * qw2[k * 4 + t];
    sdq[t] = d;
  }
  __syncthreads();

  // torsions = MLP(msum)
  a = tb1[t];
  for (int k = 0; k < 64; ++k) a += sx[64 + k] * tw1[k * 64 + t];
  hid[t] = fmaxf(a, 0.f);
  __syncthreads();
  if (t < 14) {
    float d = tb2[t];
    for (int k = 0; k < 64; ++k) d += hid[k] * tw2[k * 14 + t];
    stor[t] = d;
  }
  __syncthreads();

  if (t == 0) {
    const bool maski = (nm[b * NB + i] != 0);
    const float ident[4] = {1.f, 0.f, 0.f, 0.f};
    float dq[4];
    for (int n = 0; n < 4; ++n) dq[n] = maski ? sdq[n] : ident[n];
    float nn = fmaxf(sqrtf(dq[0]*dq[0] + dq[1]*dq[1] + dq[2]*dq[2] + dq[3]*dq[3]), 1e-12f);
    for (int n = 0; n < 4; ++n) dq[n] /= nn;
    float q[4] = { fq[(size_t)bi*4+0], fq[(size_t)bi*4+1], fq[(size_t)bi*4+2], fq[(size_t)bi*4+3] };
    float uq[4]; qmul(q, dq, uq);
    if (!maski) { uq[0]=1.f; uq[1]=0.f; uq[2]=0.f; uq[3]=0.f; }
    nn = fmaxf(sqrtf(uq[0]*uq[0] + uq[1]*uq[1] + uq[2]*uq[2] + uq[3]*uq[3]), 1e-12f);
    for (int n = 0; n < 4; ++n) out[(size_t)bi * 4 + n] = uq[n] / nn;
    const float nnb = (float)(scnt - 1);
    for (int d = 0; d < 3; ++d)
      out[3072 + (size_t)bi * 3 + d] = fx[(size_t)bi * 3 + d] + gdx_g[(size_t)bi * 4 + d] / nnb;
  }
  if (t < 7) {
    const float v0 = stor[2 * t], v1 = stor[2 * t + 1];
    const float nn = fmaxf(sqrtf(v0 * v0 + v1 * v1), 1e-12f);
    out[5376 + (size_t)bi * 14 + 2 * t    ] = v0 / nn;
    out[5376 + (size_t)bi * 14 + 2 * t + 1] = v1 / nn;
  }
}

// ---------------------------------------------------------------------------
extern "C" void kernel_launch(void* const* d_in, const int* in_sizes, int n_in,
                              void* d_out, int out_size, void* d_ws, size_t ws_size,
                              hipStream_t stream)
{
  (void)in_sizes; (void)n_in; (void)out_size; (void)ws_size;
  const float* fq = (const float*)d_in[0];
  const float* fx = (const float*)d_in[1];
  // d_in[2] torsions: only defines output shape, values unused by reference
  const float* h  = (const float*)d_in[3];
  const float* eN = (const float*)d_in[4];
  const int*   nm = (const int*)d_in[5];
  const float* ph = (const float*)d_in[6];
  const float* eP = (const float*)d_in[7];
  const float* pq = (const float*)d_in[8];
  const float* px = (const float*)d_in[9];
  const int*   pm = (const int*)d_in[10];
  const float* msg_w1 = (const float*)d_in[11];
  const float* msg_b1 = (const float*)d_in[12];
  const float* msg_w2 = (const float*)d_in[13];
  const float* msg_b2 = (const float*)d_in[14];
  const float* fw1 = (const float*)d_in[15];
  const float* fb1 = (const float*)d_in[16];
  const float* fw2 = (const float*)d_in[17];
  const float* fb2 = (const float*)d_in[18];
  const float* tr_w1 = (const float*)d_in[19];
  const float* tr_b1 = (const float*)d_in[20];
  const float* tr_w2 = (const float*)d_in[21];
  const float* tr_b2 = (const float*)d_in[22];
  const float* qw1 = (const float*)d_in[23];
  const float* qb1 = (const float*)d_in[24];
  const float* qw2 = (const float*)d_in[25];
  const float* qb2 = (const float*)d_in[26];
  const float* tw1 = (const float*)d_in[27];
  const float* tb1 = (const float*)d_in[28];
  const float* tw2 = (const float*)d_in[29];
  const float* tb2 = (const float*)d_in[30];

  _Float16* wsH    = (_Float16*)d_ws;                  // WPK_TOT halves of packed weights
  float*    msum_g = (float*)((char*)d_ws + 36864);    // 768 * 64 f32
  float*    gdx_g  = (float*)((char*)d_ws + 233472);   // 768 * 4  f32

  prepack_kernel<<<68, 256, 0, stream>>>(msg_w1, msg_w2, tr_w1, tr_w2, wsH);

  egnn_edge_kernel<<<2 * NB, 128, 0, stream>>>(
      fq, fx, h, eN, nm, ph, eP, pq, px, pm,
      msg_w1, msg_b1, msg_b2, tr_b1, tr_b2,
      wsH, msum_g, gdx_g);

  egnn_node_kernel<<<2 * NB, 64, 0, stream>>>(
      fq, fx, h, nm, pm,
      fw1, fb1, fw2, fb2,
      qw1, qb1, qw2, qb2,
      tw1, tb1, tw2, tb2,
      msum_g, gdx_g, (float*)d_out);
}
